// Head_50697793962132
// MI455X (gfx1250) — matrix operand order repaired and checked by reading, not compiled, not run on hardware
//
#include <hip/hip_runtime.h>
#include <hip/hip_bf16.h>

typedef _Float16 v16h __attribute__((ext_vector_type(16)));
typedef _Float16 v8h  __attribute__((ext_vector_type(8)));
typedef _Float16 v4h  __attribute__((ext_vector_type(4)));
typedef float    v8f  __attribute__((ext_vector_type(8)));
typedef float    v4f  __attribute__((ext_vector_type(4)));

#define N_EMBD 384
#define HS     64
#define T_SEQ  2048
#define B_BATCH 8
#define WT_STRIDE (HS * N_EMBD)   // 24576 halves per matrix

// ---------------------------------------------------------------------------
// Kernel 0: one-time transpose+convert of the three 384x64 fp32 weights into
// f16 Wt[mat][n][k] (k contiguous) so WMMA B-fragments become b128 loads.
// 144 KB total -> permanently L2-resident on a 192 MB L2.
// ---------------------------------------------------------------------------
__global__ __launch_bounds__(128) void wconv_kernel(
    const float* __restrict__ Wq, const float* __restrict__ Wk,
    const float* __restrict__ Wv, _Float16* __restrict__ wt)
{
    const int id  = blockIdx.x;          // 0..191
    const int mat = id >> 6;             // /64
    const int n   = id & 63;
    const float* __restrict__ W = (mat == 0) ? Wq : (mat == 1) ? Wk : Wv;
    for (int k = threadIdx.x; k < N_EMBD; k += 128)
        wt[mat * WT_STRIDE + n * N_EMBD + k] = (_Float16)W[(size_t)k * HS + n];
}

// ---------------------------------------------------------------------------
// Kernel 1: QKV projection.  One block = 16 rows of x; 4 waves, wave w owns
// N-tile w.  The 16x384 x tile is one contiguous 6144-float block: staged to
// LDS as f16 via b128 loads + pk-converts + b64 DS stores.  Weight
// B-fragments are contiguous v16h loads from Wt.  Q,K stored row-major f16;
// V stored TRANSPOSED (Vt[b][n][t]) so attention PV B-fragments are
// contiguous too.
// ---------------------------------------------------------------------------
__global__ __launch_bounds__(128) void qkv_proj_kernel(
    const float* __restrict__ x, const _Float16* __restrict__ wt,
    _Float16* __restrict__ qws, _Float16* __restrict__ kws,
    _Float16* __restrict__ vtws)
{
    __shared__ _Float16 sx[16 * N_EMBD];

    const int rowbase = blockIdx.x * 16;          // row in flattened [B*T]
    const int tid = threadIdx.x;

    {   // vectorized staging: 6144 floats = 128 thr * 12 it * 4-wide
        const float* __restrict__ xt = x + (size_t)rowbase * N_EMBD;
#pragma unroll
        for (int it = 0; it < (16 * N_EMBD) / (128 * 4); ++it) {
            const int i = (it * 128 + tid) * 4;
            const v4f v = *(const v4f*)(xt + i);
            v4h h;
            h[0] = (_Float16)v[0]; h[1] = (_Float16)v[1];
            h[2] = (_Float16)v[2]; h[3] = (_Float16)v[3];
            *(v4h*)(sx + i) = h;
        }
    }
    __syncthreads();

    const int lane = tid & 31;
    const int nt   = tid >> 5;                    // wave id == N-tile (0..3)
    const int hi   = lane >> 4;
    const int ln   = lane & 15;
    const int ncol = nt * 16 + ln;                // this lane's output column

#pragma unroll
    for (int mat = 0; mat < 3; ++mat) {
        const _Float16* __restrict__ Wrow =
            wt + mat * WT_STRIDE + (size_t)ncol * N_EMBD;
        v8f acc = {};
        for (int ks = 0; ks < N_EMBD / 32; ++ks) {
            const int k0 = ks * 32;
            v16h a;
#pragma unroll
            for (int t = 0; t < 16; ++t) {        // A frag (ds_load_b128 x2)
                int kk = k0 + (hi << 4) + t;
                a[t] = sx[ln * N_EMBD + kk];
            }
            const v16h b = *(const v16h*)(Wrow + k0 + (hi << 4));
            acc = __builtin_amdgcn_wmma_f32_16x16x32_f16(
                false, a, false, b, (short)0, acc, false, false);
        }
        if (mat == 0 || mat == 1) {
            _Float16* __restrict__ o = (mat == 0) ? qws : kws;
#pragma unroll
            for (int r = 0; r < 8; ++r) {
                int row = rowbase + r + 8 * hi;
                o[(size_t)row * HS + ncol] = (_Float16)acc[r];
            }
        } else {                                   // V: transposed store
#pragma unroll
            for (int r = 0; r < 8; ++r) {
                int row = rowbase + r + 8 * hi;
                int bb  = row >> 11;               // /T_SEQ
                int tt  = row & (T_SEQ - 1);
                vtws[((size_t)bb * HS + ncol) * T_SEQ + tt] = (_Float16)acc[r];
            }
        }
    }
}

// ---------------------------------------------------------------------------
// Kernel 2: fused causal flash attention.  One wave per 16-query tile;
// walks 32-key tiles up to the diagonal.  All WMMA operand loads are
// contiguous b128s; P goes C-frag -> A-frag through 1 KB LDS (single-wave
// barrier -> S_NOP but orders DS ops for the compiler).
// ---------------------------------------------------------------------------
__global__ __launch_bounds__(32) void attn_kernel(
    const _Float16* __restrict__ qws, const _Float16* __restrict__ kws,
    const _Float16* __restrict__ vtws, float* __restrict__ out)
{
    __shared__ _Float16 sp[16 * 32];

    const int b  = blockIdx.x >> 7;               // /128 tiles per batch
    const int qt = blockIdx.x & 127;
    const int qb = qt * 16;
    const size_t base = (size_t)b * T_SEQ;

    const int lane = threadIdx.x & 31;
    const int hi = lane >> 4;
    const int ln = lane & 15;

    const _Float16 hscale = (_Float16)0.05103103630798287f;  // 1/sqrt(384)

    // Q fragments (16x64 -> two 16x32 A frags), scale folded in.
    v16h qa[2];
#pragma unroll
    for (int ks = 0; ks < 2; ++ks) {
        const _Float16* qp = qws + (base + qb + ln) * HS + ks * 32 + (hi << 4);
        const v8h lo  = *(const v8h*)qp;           // K = ks*32 + 16*hi + 0..7
        const v8h hi8 = *(const v8h*)(qp + 8);     // K = ks*32 + 16*hi + 8..15
#pragma unroll
        for (int t = 0; t < 8; ++t) {
            qa[ks][t]     = lo[t]  * hscale;
            qa[ks][t + 8] = hi8[t] * hscale;
        }
    }

    v8f acc[4] = {};
    float m_run[8], l_run[8];
#pragma unroll
    for (int r = 0; r < 8; ++r) { m_run[r] = -1e30f; l_run[r] = 0.0f; }

    const int nkt = (qb + 47) >> 5;               // tiles covering keys [0, qb+15]
    for (int it = 0; it < nkt; ++it) {
        const int kb = it * 32;

        // ---- S = Q K^T for 32 keys (two 16-key C fragments) ----
        v8f s[2] = {};
#pragma unroll
        for (int j = 0; j < 2; ++j) {
#pragma unroll
            for (int ks = 0; ks < 2; ++ks) {
                const v16h bk = *(const v16h*)(kws +
                    (base + kb + 16 * j + ln) * HS + ks * 32 + (hi << 4));
                s[j] = __builtin_amdgcn_wmma_f32_16x16x32_f16(
                    false, qa[ks], false, bk, (short)0, s[j], false, false);
            }
        }

        // ---- causal mask (diagonal tiles only) ----
        if (kb + 32 > qb) {
#pragma unroll
            for (int j = 0; j < 2; ++j) {
                int key = kb + 16 * j + ln;
#pragma unroll
                for (int r = 0; r < 8; ++r) {
                    int row = qb + r + 8 * hi;
                    if (key > row) s[j][r] = -1e30f;
                }
            }
        }

        // ---- online softmax; stage P to LDS for layout change ----
#pragma unroll
        for (int r = 0; r < 8; ++r) {
            float rm = fmaxf(s[0][r], s[1][r]);
            rm = fmaxf(rm, __shfl_xor(rm, 1, 32));
            rm = fmaxf(rm, __shfl_xor(rm, 2, 32));
            rm = fmaxf(rm, __shfl_xor(rm, 4, 32));
            rm = fmaxf(rm, __shfl_xor(rm, 8, 32));
            float mnew = fmaxf(m_run[r], rm);
            float corr = __expf(m_run[r] - mnew);
            float p0 = __expf(s[0][r] - mnew);
            float p1 = __expf(s[1][r] - mnew);
            float rs = p0 + p1;
            rs += __shfl_xor(rs, 1, 32);
            rs += __shfl_xor(rs, 2, 32);
            rs += __shfl_xor(rs, 4, 32);
            rs += __shfl_xor(rs, 8, 32);
            l_run[r] = l_run[r] * corr + rs;
            m_run[r] = mnew;
#pragma unroll
            for (int n2 = 0; n2 < 4; ++n2) acc[n2][r] *= corr;

            sp[(r + 8 * hi) * 32 + ln]      = (_Float16)p0;
            sp[(r + 8 * hi) * 32 + 16 + ln] = (_Float16)p1;
        }
        __syncthreads();                          // single-wave: S_NOP

        v16h pa;                                  // P as 16x32 A fragment
#pragma unroll
        for (int t = 0; t < 16; ++t) {
            int kk = (hi << 4) + t;
            pa[t] = sp[ln * 32 + kk];
        }
        __syncthreads();

        // ---- O += P V  (V transposed: contiguous B fragments) ----
#pragma unroll
        for (int n2 = 0; n2 < 4; ++n2) {
            const v16h bv = *(const v16h*)(vtws +
                ((size_t)b * HS + n2 * 16 + ln) * T_SEQ + kb + (hi << 4));
            acc[n2] = __builtin_amdgcn_wmma_f32_16x16x32_f16(
                false, pa, false, bv, (short)0, acc[n2], false, false);
        }
    }

    // ---- epilogue: normalize and store fp32 ----
#pragma unroll
    for (int r = 0; r < 8; ++r) {
        float inv = 1.0f / l_run[r];
        int row = qb + r + 8 * hi;
#pragma unroll
        for (int n2 = 0; n2 < 4; ++n2)
            out[(base + row) * HS + n2 * 16 + ln] = acc[n2][r] * inv;
    }
}

// ---------------------------------------------------------------------------
extern "C" void kernel_launch(void* const* d_in, const int* in_sizes, int n_in,
                              void* d_out, int out_size, void* d_ws,
                              size_t ws_size, hipStream_t stream)
{
    const float* x  = (const float*)d_in[0];
    const float* Wq = (const float*)d_in[1];
    const float* Wk = (const float*)d_in[2];
    const float* Wv = (const float*)d_in[3];
    float* out = (float*)d_out;

    const size_t qsz = (size_t)B_BATCH * T_SEQ * HS;   // 1M elems
    _Float16* qws  = (_Float16*)d_ws;                  // 2 MB
    _Float16* kws  = qws + qsz;                        // 2 MB
    _Float16* vtws = kws + qsz;                        // 2 MB (transposed V)
    _Float16* wt   = vtws + qsz;                       // 144 KB (f16 weights)

    wconv_kernel<<<3 * HS, 128, 0, stream>>>(Wq, Wk, Wv, wt);
    qkv_proj_kernel<<<B_BATCH * T_SEQ / 16, 128, 0, stream>>>(x, wt, qws, kws, vtws);
    attn_kernel<<<B_BATCH * (T_SEQ / 16), 32, 0, stream>>>(qws, kws, vtws, out);
}
